// LSTM_85349590106282
// MI455X (gfx1250) — matrix operand order repaired and checked by reading, not compiled, not run on hardware
//
#include <hip/hip_runtime.h>
#include <hip/hip_bf16.h>

// ---------------------------------------------------------------------------
// Bidirectional LSTM (T=64, N=8192, EMB=128, H=32) fused for gfx1250 (MI455X).
// Kernel 1: embedding gather + input projection + recurrent scan, all WMMA bf16
//           with fp32 accumulation; h-state in LDS, c-state in VGPRs.
//           Activations use gfx1250 hardware v_tanh_f32.
// Kernel 2: max-pool over batch + tiny output GEMM.
// Workspace: hs buffer [T][N][2H] fp32 = 128 MB.
// ---------------------------------------------------------------------------

typedef __bf16 bf16_t;
typedef __attribute__((ext_vector_type(16))) __bf16 v16bf;
typedef __attribute__((ext_vector_type(8)))  __bf16 v8bf;
typedef __attribute__((ext_vector_type(8)))  float  v8f;

#define T_STEPS 64
#define NBATCH  8192
#define EMBD    128
#define HID     32
#define GATES   128   // 4*H
#define ROWS    128   // batch rows per block
#define PAD_IDX 1

#if defined(__has_builtin) && __has_builtin(__builtin_amdgcn_tanhf)
// gfx1250 hardware transcendental: v_tanh_f32
__device__ __forceinline__ float tanh_(float x) {
  return __builtin_amdgcn_tanhf(x);
}
__device__ __forceinline__ float sigmoid_(float x) {
  return 0.5f * __builtin_amdgcn_tanhf(0.5f * x) + 0.5f;
}
#else
__device__ __forceinline__ float tanh_(float x) {
  x = fminf(10.0f, fmaxf(-10.0f, x));
  float e = __expf(2.0f * x);
  return (e - 1.0f) * __builtin_amdgcn_rcpf(e + 1.0f);
}
__device__ __forceinline__ float sigmoid_(float x) {
  return __builtin_amdgcn_rcpf(1.0f + __expf(-x));
}
#endif

// Build a 16-wide bf16 fragment from two 8-element (16B) LDS chunks.
__device__ __forceinline__ v16bf ld_frag(const bf16_t* lo, const bf16_t* hi) {
  v8bf a = *(const v8bf*)lo;
  v8bf b = *(const v8bf*)hi;
  v16bf r;
#pragma unroll
  for (int i = 0; i < 8; ++i) { r[i] = a[i]; r[8 + i] = b[i]; }
  return r;
}

__global__ __launch_bounds__(256) void lstm_fused(
    const int*   __restrict__ x,       // [T][N]
    const float* __restrict__ emb_w,   // [VOCAB][EMB]
    const float* __restrict__ Wih_f,   // [4H][EMB]
    const float* __restrict__ Whh_f,   // [4H][H]
    const float* __restrict__ bih_f, const float* __restrict__ bhh_f,
    const float* __restrict__ Wih_r, const float* __restrict__ Whh_r,
    const float* __restrict__ bih_r, const float* __restrict__ bhh_r,
    float* __restrict__ d_hs)          // [T][N][2H]
{
  // ---- LDS staging (total ~132 KB of the 320 KB WGP pool) ----
  __shared__ __align__(16) bf16_t sWih[2][GATES][EMBD];  // 64 KB
  __shared__ __align__(16) bf16_t sWhh[2][GATES][HID];   // 16 KB
  __shared__ float  sBias[2][GATES];                     // 1 KB
  __shared__ int    sTok[ROWS];
  __shared__ __align__(16) bf16_t sE[ROWS][EMBD];        // 32 KB
  __shared__ __align__(16) bf16_t sH[2][ROWS][HID];      // 16 KB

  const int tid  = threadIdx.x;
  const int n0   = blockIdx.x * ROWS;
  const int w    = tid >> 5;        // wave id 0..7
  const int lane = tid & 31;
  const int ug   = w & 1;           // unit-group: units [ug*16, ug*16+16)
  const int rp   = w >> 1;          // row-pair: rows [rp*32, rp*32+32)
  const int hsel = lane >> 4;       // half-wave select
  const int ln   = lane & 15;

  // Stage weights (fp32 -> bf16) and combined biases.
  for (int i = tid; i < GATES * EMBD; i += 256) {
    sWih[0][0][i] = (bf16_t)Wih_f[i];
    sWih[1][0][i] = (bf16_t)Wih_r[i];
  }
  for (int i = tid; i < GATES * HID; i += 256) {
    sWhh[0][0][i] = (bf16_t)Whh_f[i];
    sWhh[1][0][i] = (bf16_t)Whh_r[i];
  }
  for (int i = tid; i < GATES; i += 256) {
    sBias[0][i] = bih_f[i] + bhh_f[i];
    sBias[1][i] = bih_r[i] + bhh_r[i];
  }
  for (int i = tid; i < 2 * ROWS * HID; i += 256)
    ((bf16_t*)sH)[i] = (bf16_t)0.0f;
  __syncthreads();

  // Per-lane bias for its 4 gate columns (col = q*32 + ug*16 + ln).
  float biasv[2][4];
#pragma unroll
  for (int d = 0; d < 2; ++d)
#pragma unroll
    for (int q = 0; q < 4; ++q)
      biasv[d][q] = sBias[d][q * 32 + ug * 16 + ln];

  // Cell state lives in registers for all 64 steps.
  float cst[2][2][8];
#pragma unroll
  for (int d = 0; d < 2; ++d)
#pragma unroll
    for (int rt = 0; rt < 2; ++rt)
#pragma unroll
      for (int i = 0; i < 8; ++i) cst[d][rt][i] = 0.0f;

  for (int t = 0; t < T_STEPS; ++t) {
#pragma unroll
    for (int d = 0; d < 2; ++d) {
      const int tt = d ? (T_STEPS - 1 - t) : t;   // reverse dir walks backwards

      // ---- tokens for this timestep / batch tile ----
      if (tid < ROWS) sTok[tid] = x[tt * NBATCH + n0 + tid];
      __syncthreads();

      // ---- embedding gather -> bf16 LDS tile (pad row forced to zero) ----
      for (int i = tid; i < ROWS * (EMBD / 4); i += 256) {
        const int r = i >> 5, c4 = i & 31;
        const int tok = sTok[r];
        float4 v = make_float4(0.f, 0.f, 0.f, 0.f);
        if (tok != PAD_IDX)
          v = ((const float4*)(emb_w + (size_t)tok * EMBD))[c4];
        bf16_t* dst = &sE[r][c4 * 4];
        dst[0] = (bf16_t)v.x; dst[1] = (bf16_t)v.y;
        dst[2] = (bf16_t)v.z; dst[3] = (bf16_t)v.w;
      }
      __syncthreads();

      // ---- gates = bias + E@Wih^T + h@Whh^T  via v_wmma_f32_16x16x32_bf16,
      //      then activations + c/h update per row-tile (keeps acc lifetimes
      //      short); sH rewrite is deferred until after the barrier. ----
      float hval[2][8];

#pragma unroll
      for (int rt = 0; rt < 2; ++rt) {
        const int arow = (rp * 2 + rt) * 16 + ln;  // A-matrix row for this lane

        // A fragments from the embedding tile: K contiguous 0..15 (+hsel*16).
        v16bf ae[4];
#pragma unroll
        for (int kk = 0; kk < 4; ++kk) {
          const bf16_t* er = &sE[arow][kk * 32 + hsel * 16];
          ae[kk] = ld_frag(er, er + 8);
        }
        // A fragment from h-state (K = 32 hidden units).
        const bf16_t* hr = &sH[d][arow][hsel * 16];
        const v16bf ah = ld_frag(hr, hr + 8);

        v8f acc[4];
#pragma unroll
        for (int q = 0; q < 4; ++q) {
          const int g = q * 32 + ug * 16 + ln;     // gate column (B lane col)
          v8f a8;
#pragma unroll
          for (int i = 0; i < 8; ++i) a8[i] = biasv[d][q];
#pragma unroll
          for (int kk = 0; kk < 4; ++kk) {
            const bf16_t* wr = &sWih[d][g][kk * 32 + hsel * 16];
            const v16bf b = ld_frag(wr, wr + 8);
            a8 = __builtin_amdgcn_wmma_f32_16x16x32_bf16(
                false, ae[kk], false, b, (short)0, a8, false, false);
          }
          const bf16_t* wh = &sWhh[d][g][hsel * 16];
          const v16bf bh = ld_frag(wh, wh + 8);
          a8 = __builtin_amdgcn_wmma_f32_16x16x32_bf16(
              false, ah, false, bh, (short)0, a8, false, false);
          acc[q] = a8;
        }

        // Gate nonlinearities + cell update (register-only, v_tanh_f32).
        const int rbase = (rp * 2 + rt) * 16 + hsel * 8;  // C-tile row base
        const int u = ug * 16 + ln;                        // hidden unit
#pragma unroll
        for (int i = 0; i < 8; ++i) {
          const float iv = sigmoid_(acc[0][i]);
          const float fv = sigmoid_(acc[1][i]);
          const float gv = tanh_(acc[2][i]);
          const float ov = sigmoid_(acc[3][i]);
          const float cc = fv * cst[d][rt][i] + iv * gv;
          cst[d][rt][i] = cc;
          const float hv = ov * tanh_(cc);
          hval[rt][i] = hv;
          d_hs[((size_t)tt * NBATCH + n0 + rbase + i) * 64 + d * 32 + u] = hv;
        }
      }

      __syncthreads();   // all reads of sE / sH[d] complete before rewrite

#pragma unroll
      for (int rt = 0; rt < 2; ++rt) {
        const int rbase = (rp * 2 + rt) * 16 + hsel * 8;
        const int u = ug * 16 + ln;
#pragma unroll
        for (int i = 0; i < 8; ++i)
          sH[d][rbase + i][u] = (bf16_t)hval[rt][i];
      }
      // next-iteration barriers (token/emb) order sH writes before reads
    }
  }
}

// ---- max over batch (dim=1) then [T,64] @ [64,2] + b ----
__global__ __launch_bounds__(256) void pool_out(
    const float* __restrict__ hs,    // [T][N][64]
    const float* __restrict__ Wout,  // [2][64]
    const float* __restrict__ bout,  // [2]
    float* __restrict__ out)         // [T][2]
{
  __shared__ float red[4][64];
  __shared__ float pooled[64];
  const int t = blockIdx.x;
  const int tid = threadIdx.x;
  const int col = tid & 63, chunk = tid >> 6;

  float m = -3.402823466e38f;
  const float* base = hs + (size_t)t * NBATCH * 64 + col;
  for (int r = chunk; r < NBATCH; r += 4)
    m = fmaxf(m, base[(size_t)r * 64]);
  red[chunk][col] = m;
  __syncthreads();
  if (tid < 64)
    pooled[tid] = fmaxf(fmaxf(red[0][tid], red[1][tid]),
                        fmaxf(red[2][tid], red[3][tid]));
  __syncthreads();
  if (tid < 2) {
    float s = bout[tid];
#pragma unroll
    for (int c = 0; c < 64; ++c) s += pooled[c] * Wout[tid * 64 + c];
    out[t * 2 + tid] = s;
  }
}

extern "C" void kernel_launch(void* const* d_in, const int* in_sizes, int n_in,
                              void* d_out, int out_size, void* d_ws, size_t ws_size,
                              hipStream_t stream) {
  const int*   x     = (const int*)  d_in[0];
  const float* emb_w = (const float*)d_in[1];
  const float* Wih_f = (const float*)d_in[2];
  const float* Whh_f = (const float*)d_in[3];
  const float* bih_f = (const float*)d_in[4];
  const float* bhh_f = (const float*)d_in[5];
  const float* Wih_r = (const float*)d_in[6];
  const float* Whh_r = (const float*)d_in[7];
  const float* bih_r = (const float*)d_in[8];
  const float* bhh_r = (const float*)d_in[9];
  const float* Wout  = (const float*)d_in[10];
  const float* bout  = (const float*)d_in[11];

  float* hs = (float*)d_ws;  // requires T*N*2H*4 = 128 MB workspace

  lstm_fused<<<NBATCH / ROWS, 256, 0, stream>>>(
      x, emb_w, Wih_f, Whh_f, bih_f, bhh_f,
      Wih_r, Whh_r, bih_r, bhh_r, hs);
  pool_out<<<T_STEPS, 256, 0, stream>>>(hs, Wout, bout, (float*)d_out);
}
